// LinearStateSpaceModel_61649960566906
// MI455X (gfx1250) — compile-verified, matrix operand order checked by reading, not run on hardware
//
#include <hip/hip_runtime.h>

// ---------------------------------------------------------------------------
// Types for CDNA5 WMMA (gfx1250, wave32)
// ---------------------------------------------------------------------------
typedef __attribute__((ext_vector_type(16))) __bf16 v16bf;
typedef __attribute__((ext_vector_type(8)))  float  v8f;

#define WMMA_BF16(a, b, c) \
  __builtin_amdgcn_wmma_f32_16x16x32_bf16(false, (a), false, (b), (short)0, (c), false, false)

// Problem constants
static constexpr int B_SZ = 8;
static constexpr int T    = 16;
static constexpr int P    = 256;
static constexpr int D    = 1024;
static constexpr int S    = 1024;
static constexpr int M_ALL = B_SZ * T * P;   // 32768 rows
static constexpr int KDIM  = 1024;
static constexpr int NDIM  = 1024;

// GEMM tiling
static constexpr int BM = 128;
static constexpr int BN = 256;
static constexpr int KC = 64;

// ---------------------------------------------------------------------------
// CDNA5 async global->LDS copy (ISA ch.10: GLOBAL_LOAD_ASYNC_TO_LDS_B128,
// tracked with ASYNCcnt). LDS offset = low 32 bits of the generic LDS pointer.
// ---------------------------------------------------------------------------
__device__ __forceinline__ void async_ld16(void* lds_ptr, const void* gptr) {
  unsigned lds_off = (unsigned)(unsigned long long)lds_ptr;
  unsigned long long ga = (unsigned long long)gptr;
  asm volatile("global_load_async_to_lds_b128 %0, %1, off"
               :: "v"(lds_off), "v"(ga) : "memory");
}
__device__ __forceinline__ void wait_async0() {
  asm volatile("s_wait_asynccnt 0x0" ::: "memory");
}

// ---------------------------------------------------------------------------
// Fragment loader: 16-bit A/B WMMA layout (ISA 7.12.2).
// lane&15 -> row within tile; lane>>4 selects K-halves {0..7,16..23} vs {8..15,24..31}
// ---------------------------------------------------------------------------
__device__ __forceinline__ v16bf ld_frag_bf16(const __bf16* base, int row_base, int koff,
                                              int lane, int ldk) {
  const int l15 = lane & 15;
  const int k0  = koff + ((lane >> 4) << 3);
  const __bf16* p = base + (size_t)(row_base + l15) * ldk + k0;
  union { v16bf v; uint4 u[2]; } f;
  f.u[0] = *reinterpret_cast<const uint4*>(p);        // K = k0..k0+7
  f.u[1] = *reinterpret_cast<const uint4*>(p + 16);   // K = k0+16..k0+23
  return f.v;
}

// ---------------------------------------------------------------------------
// Kernel 1: xb = bf16(x + pos_embedding)   (pos broadcast over batch)
// ---------------------------------------------------------------------------
__global__ __launch_bounds__(256) void k_addpos(const float* __restrict__ x,
                                                const float* __restrict__ pos,
                                                __bf16* __restrict__ xb) {
  const size_t nd    = (size_t)(T * P) * D;
  const size_t i     = ((size_t)blockIdx.x * blockDim.x + threadIdx.x) * 4;
  const size_t total = (size_t)B_SZ * nd;
  if (i >= total) return;
  const size_t pidx = i % nd;
  float4 xv = *reinterpret_cast<const float4*>(x + i);
  float4 pv = *reinterpret_cast<const float4*>(pos + pidx);
  union { uint2 u; __bf16 h[4]; } o;
  o.h[0] = (__bf16)(xv.x + pv.x);
  o.h[1] = (__bf16)(xv.y + pv.y);
  o.h[2] = (__bf16)(xv.z + pv.z);
  o.h[3] = (__bf16)(xv.w + pv.w);
  *reinterpret_cast<uint2*>(xb + i) = o.u;
}

// ---------------------------------------------------------------------------
// Kernel 2a: bf16 copy-convert (W_in, W_out)
// ---------------------------------------------------------------------------
__global__ __launch_bounds__(256) void k_cvt_copy(const float* __restrict__ w,
                                                  __bf16* __restrict__ wb, int n) {
  int i = (blockIdx.x * 256 + threadIdx.x) * 4;
  if (i >= n) return;
  float4 v = *reinterpret_cast<const float4*>(w + i);
  union { uint2 u; __bf16 h[4]; } o;
  o.h[0] = (__bf16)v.x; o.h[1] = (__bf16)v.y; o.h[2] = (__bf16)v.z; o.h[3] = (__bf16)v.w;
  *reinterpret_cast<uint2*>(wb + i) = o.u;
}

// ---------------------------------------------------------------------------
// Kernel 2b: bf16 transpose-convert (Bm -> Bm^T, A -> A^T), 1024x1024
// ---------------------------------------------------------------------------
__global__ __launch_bounds__(256) void k_cvt_trans(const float* __restrict__ w,
                                                   __bf16* __restrict__ wt) {
  int i = blockIdx.x * 256 + threadIdx.x;     // output element index, row-major [n][k]
  int n = i >> 10, k = i & 1023;
  wt[i] = (__bf16)w[k * 1024 + n];
}

// ---------------------------------------------------------------------------
// Main GEMM: C[M,N] = A[M,K](bf16) * Bmat[N,K](bf16)^T  (+bias) (+row-broadcast U)
// Double-buffered LDS; staging via async global->LDS (sync VALU path when ADD_U
// must fold U into the A tile).
// ---------------------------------------------------------------------------
template <bool ADD_U, bool HAS_BIAS, bool OUT_F32>
__global__ __launch_bounds__(256) void k_gemm(const __bf16* __restrict__ Amat,
                                              const __bf16* __restrict__ Bmat,
                                              const float*  __restrict__ bias,
                                              const float*  __restrict__ U,
                                              float*  __restrict__ Cf,
                                              __bf16* __restrict__ Cb,
                                              int M, int N, int K) {
  __shared__ __align__(16) __bf16 sA[2][BM * KC];   // 2 x 16 KB
  __shared__ __align__(16) __bf16 sB[2][BN * KC];   // 2 x 32 KB

  const int tid  = threadIdx.x;
  const int lane = tid & 31;
  const int wid  = tid >> 5;          // 8 waves
  const int wm   = wid & 1;           // 2 in M
  const int wn   = wid >> 1;          // 4 in N

  const int bn    = blockIdx.x & ((N / BN) - 1);
  const int bm    = blockIdx.x / (N / BN);
  const int mbase = bm * BM;
  const int nbase = bn * BN;
  const int urow  = mbase >> 8;       // (b*T+t): BM=128 tiles never cross P=256 rows

  auto stageA = [&](int buf, int kb) {
#pragma unroll
    for (int c = tid; c < (BM * KC) / 8; c += 256) {
      const int row = c >> 3;
      const int c8  = c & 7;
      const __bf16* gp = Amat + (size_t)(mbase + row) * K + kb + c8 * 8;
      if (ADD_U) {
        union { uint4 u; __bf16 h[8]; } v;
        v.u = *reinterpret_cast<const uint4*>(gp);
        const float* up = U + (size_t)urow * K + kb + c8 * 8;
#pragma unroll
        for (int j = 0; j < 8; ++j) v.h[j] = (__bf16)((float)v.h[j] + up[j]);
        *reinterpret_cast<uint4*>(&sA[buf][row * KC + c8 * 8]) = v.u;
      } else {
        async_ld16(&sA[buf][row * KC + c8 * 8], gp);
      }
    }
  };
  auto stageB = [&](int buf, int kb) {
#pragma unroll
    for (int c = tid; c < (BN * KC) / 8; c += 256) {
      const int row = c >> 3;
      const int c8  = c & 7;
      async_ld16(&sB[buf][row * KC + c8 * 8],
                 Bmat + (size_t)(nbase + row) * K + kb + c8 * 8);
    }
  };

  v8f acc[4][4];
#pragma unroll
  for (int i = 0; i < 4; ++i)
#pragma unroll
    for (int j = 0; j < 4; ++j) acc[i][j] = (v8f)0.0f;

  stageA(0, 0);
  stageB(0, 0);

  int cur = 0;
  for (int kb = 0; kb < K; kb += KC) {
    wait_async0();        // own async copies into sA[cur]/sB[cur] landed
    __syncthreads();      // everyone's copies landed (also flushes ds stores for ADD_U)

    if (kb + KC < K) {    // prefetch next tiles into the other buffer
      stageA(cur ^ 1, kb + KC);
      stageB(cur ^ 1, kb + KC);
    }

    const __bf16* a_tile = sA[cur];
    const __bf16* b_tile = sB[cur];
#pragma unroll
    for (int kk = 0; kk < KC; kk += 32) {
      v16bf af[4], bfv[4];
#pragma unroll
      for (int i = 0; i < 4; ++i) af[i]  = ld_frag_bf16(a_tile, wm * 64 + i * 16, kk, lane, KC);
#pragma unroll
      for (int j = 0; j < 4; ++j) bfv[j] = ld_frag_bf16(b_tile, wn * 64 + j * 16, kk, lane, KC);
#pragma unroll
      for (int i = 0; i < 4; ++i)
#pragma unroll
        for (int j = 0; j < 4; ++j) acc[i][j] = WMMA_BF16(af[i], bfv[j], acc[i][j]);
    }
    cur ^= 1;
  }

  // ---- epilogue: C layout lane&15=col, lane>>4 selects M-half, vgpr r = row ----
  const int l15 = lane & 15;
  const int mo  = (lane >> 4) * 8;
#pragma unroll
  for (int j = 0; j < 4; ++j) {
    const int gc = nbase + wn * 64 + j * 16 + l15;
    const float bv = HAS_BIAS ? bias[gc] : 0.0f;
#pragma unroll
    for (int i = 0; i < 4; ++i) {
      const int gr0 = mbase + wm * 64 + i * 16 + mo;
#pragma unroll
      for (int r = 0; r < 8; ++r) {
        const float v = acc[i][j][r] + bv;
        const size_t off = (size_t)(gr0 + r) * N + gc;
        if (OUT_F32) Cf[off] = v;
        else         Cb[off] = (__bf16)v;
      }
    }
  }
}

// ---------------------------------------------------------------------------
// Kernel 5: qbar[bt][s] = mean_p Q[bt][p][s]
// ---------------------------------------------------------------------------
__global__ __launch_bounds__(256) void k_meanp(const __bf16* __restrict__ Q,
                                               float* __restrict__ qbar) {
  const int bt = blockIdx.x;                                 // 0..127
  const __bf16* base = Q + (size_t)bt * P * S;
  for (int s = threadIdx.x; s < S; s += 256) {
    float acc = 0.0f;
#pragma unroll 4
    for (int p = 0; p < P; ++p) acc += (float)base[(size_t)p * S + s];
    qbar[(size_t)bt * S + s] = acc * (1.0f / (float)P);
  }
}

// ---------------------------------------------------------------------------
// Kernel 6: sequential scan (single workgroup of 512 = 16 waves, WMMA):
//   s_0 = broadcast(init);  for t: u_t = s_t @ A;  s_{t+1} = u_t + qbar_t
// 16 waves x 4 n-tiles (64 cols each) => low register pressure, no spills.
// ---------------------------------------------------------------------------
__device__ __forceinline__ v16bf ld_frag_state(const float* base, int lane, int kb) {
  const int row = lane & 15;
  const int k0  = kb + ((lane >> 4) << 3);
  const float* p = base + row * 1024 + k0;
  union { v16bf v; __bf16 h[16]; } f;
#pragma unroll
  for (int j = 0; j < 8; ++j) f.h[j]     = (__bf16)p[j];
#pragma unroll
  for (int j = 0; j < 8; ++j) f.h[8 + j] = (__bf16)p[16 + j];
  return f.v;
}

__global__ __launch_bounds__(512) void k_scan(const __bf16* __restrict__ At,
                                              const float*  __restrict__ qbar,
                                              const float*  __restrict__ init,
                                              float* __restrict__ U) {
  __shared__ __align__(16) float s_state[16 * 1024];   // rows 8..15 stay zero (64 KB)

  const int tid  = threadIdx.x;
  const int lane = tid & 31;
  const int wid  = tid >> 5;          // 16 waves, each owns 64 output columns

  for (int i = tid; i < 16 * 1024; i += 512) {
    const int r = i >> 10, n = i & 1023;
    s_state[i] = (r < 8) ? init[n] : 0.0f;
  }
  __syncthreads();

  for (int t = 0; t < T; ++t) {
    v8f acc[4];
#pragma unroll
    for (int j = 0; j < 4; ++j) acc[j] = (v8f)0.0f;

#pragma unroll 1
    for (int kb = 0; kb < S; kb += 32) {
      const v16bf a = ld_frag_state(s_state, lane, kb);
#pragma unroll
      for (int j = 0; j < 4; ++j) {
        const v16bf b = ld_frag_bf16(At, wid * 64 + j * 16, kb, lane, 1024);
        acc[j] = WMMA_BF16(a, b, acc[j]);
      }
    }
    __syncthreads();   // all waves done reading s_state
    if (lane < 16) {
#pragma unroll
      for (int j = 0; j < 4; ++j) {
        const int n = wid * 64 + j * 16 + lane;
#pragma unroll
        for (int r = 0; r < 8; ++r) {        // r = batch b
          const float u = acc[j][r];
          const size_t idx = ((size_t)(r * T + t) << 10) + n;
          U[idx] = u;
          s_state[r * 1024 + n] = u + qbar[idx];
        }
      }
    }
    __syncthreads();   // new state visible before next step
  }
}

// ---------------------------------------------------------------------------
// Host-side orchestration
// ---------------------------------------------------------------------------
extern "C" void kernel_launch(void* const* d_in, const int* in_sizes, int n_in,
                              void* d_out, int out_size, void* d_ws, size_t ws_size,
                              hipStream_t stream) {
  (void)in_sizes; (void)n_in; (void)out_size; (void)ws_size;

  const float* x     = (const float*)d_in[0];
  const float* pos   = (const float*)d_in[1];
  const float* W_in  = (const float*)d_in[2];
  const float* b_in  = (const float*)d_in[3];
  const float* W_out = (const float*)d_in[4];
  const float* b_out = (const float*)d_in[5];
  const float* A     = (const float*)d_in[6];
  const float* Bm    = (const float*)d_in[7];
  const float* init  = (const float*)d_in[8];
  float* out = (float*)d_out;

  char* ws = (char*)d_ws;
  const size_t ACT_BYTES = (size_t)M_ALL * S * 2;            // 64 MB
  __bf16* xb     = (__bf16*)(ws);                            // xpe bf16; later reused as Q
  __bf16* proj   = (__bf16*)(ws + ACT_BYTES);                // proj bf16
  __bf16* Q      = xb;                                       // GEMM2 output overwrites xb
  char* wsw      = ws + 2 * ACT_BYTES;
  __bf16* w_in_b  = (__bf16*)(wsw);
  __bf16* bmT_b   = (__bf16*)(wsw + 1 * 2097152);
  __bf16* w_out_b = (__bf16*)(wsw + 2 * 2097152);
  __bf16* at_b    = (__bf16*)(wsw + 3 * 2097152);
  float*  qbar    = (float*)(wsw + 4 * 2097152);             // 512 KB
  float*  U       = (float*)(wsw + 4 * 2097152 + 524288);    // 512 KB

  // 1) x + pos -> bf16
  {
    const size_t total = (size_t)M_ALL * D;                  // 33554432
    k_addpos<<<dim3((unsigned)(total / 4 / 256)), dim3(256), 0, stream>>>(x, pos, xb);
  }
  // 2) weights: convert / transpose-convert
  k_cvt_copy<<<dim3(1024), dim3(256), 0, stream>>>(W_in,  w_in_b,  S * D);
  k_cvt_copy<<<dim3(1024), dim3(256), 0, stream>>>(W_out, w_out_b, D * S);
  k_cvt_trans<<<dim3(4096), dim3(256), 0, stream>>>(Bm, bmT_b);
  k_cvt_trans<<<dim3(4096), dim3(256), 0, stream>>>(A,  at_b);

  const dim3 ggrid((M_ALL / BM) * (NDIM / BN));   // 1024 blocks
  const dim3 gblk(256);

  // 3) proj = xpe @ W_in^T + b_in   (bf16 out)
  k_gemm<false, true, false><<<ggrid, gblk, 0, stream>>>(
      xb, w_in_b, b_in, nullptr, nullptr, proj, M_ALL, NDIM, KDIM);

  // 4) Q = proj @ Bm  (== proj @ (Bm^T)^T)   (bf16 out, overwrites xb)
  k_gemm<false, false, false><<<ggrid, gblk, 0, stream>>>(
      proj, bmT_b, nullptr, nullptr, nullptr, Q, M_ALL, NDIM, KDIM);

  // 5) qbar = mean over p
  k_meanp<<<dim3(B_SZ * T), dim3(256), 0, stream>>>(Q, qbar);

  // 6) sequential recurrence (single workgroup, WMMA)
  k_scan<<<dim3(1), dim3(512), 0, stream>>>(at_b, qbar, init, U);

  // 7) out = (Q + broadcast U) @ W_out^T + b_out   (f32 out)
  k_gemm<true, true, true><<<ggrid, gblk, 0, stream>>>(
      Q, w_out_b, b_out, U, out, nullptr, M_ALL, NDIM, KDIM);
}
